// LRUOptimized_80161269613385
// MI455X (gfx1250) — matrix operand order repaired and checked by reading, not compiled, not run on hardware
//
#include <hip/hip_runtime.h>

// ---------------------------------------------------------------------------
// LRU layer for MI455X (gfx1250, wave32, WMMA + TDM).
//   BATCH=16, T=4096, IN=256, OUT=256, H=512
// Pipeline:
//   K0a: Xh  = bf16(X)                        [65536x256]
//   K0b: W1  = bf16([B_re;B_im])              [1024x256]
//        W2  = bf16([C_re | -C_im | D])       [256x1280]
//   K1 : U   = gamma*(Xh @ W1^T + bh)  (WMMA) -> bf16 U_re/U_im [65536x512]
//   K2 : serial scan over T, in place         -> ys overwrites U, h_N out
//   K3 : Y   = [ys_re ys_im Xh] @ W2^T + bias (WMMA, K=1280)
// Tile loads via tensor_load_to_lds (TDM, 6-arg clang-23 form), double-
// buffered, DMA-applied LDS padding to a stride-40-ushort layout.
// ---------------------------------------------------------------------------

typedef __attribute__((ext_vector_type(16))) __bf16 v16bf;
typedef __attribute__((ext_vector_type(8)))  float  v8f;
typedef unsigned int uint4v __attribute__((ext_vector_type(4)));
typedef int          int8v  __attribute__((ext_vector_type(8)));
typedef int          int4v  __attribute__((ext_vector_type(4)));

#define BATCH    16
#define TLEN     4096
#define IN_DIM   256
#define OUT_DIM  256
#define HDIM     512
#define MROWS    (BATCH * TLEN)      // 65536

#define BM 128
#define BN 64
#define BK 32
#define LDS_STRIDE 40                // ushort stride: 32 data + 8 pad = 80 B rows

#if defined(__has_builtin)
# if __has_builtin(__builtin_amdgcn_tensor_load_to_lds) && \
     __has_builtin(__builtin_amdgcn_s_wait_tensorcnt)
#  define HAVE_TDM 1
# endif
#endif
#ifndef HAVE_TDM
# define HAVE_TDM 0
#endif

// ---------------- bf16 helpers -------------------------------------------
__device__ __forceinline__ unsigned short f32_to_bf16_rne(float f) {
  union { float f; unsigned u; } x; x.f = f;
  unsigned r = x.u + 0x7FFFu + ((x.u >> 16) & 1u);
  return (unsigned short)(r >> 16);
}
__device__ __forceinline__ unsigned short f32_to_bf16_trunc(float f) {
  union { float f; unsigned u; } x; x.f = f;
  return (unsigned short)(x.u >> 16);
}
__device__ __forceinline__ float bf16_u_to_f32(unsigned short s) {
  union { unsigned u; float f; } x; x.u = ((unsigned)s) << 16;
  return x.f;
}
// pack two f32 -> dword of 2 bf16 (truncation) in ONE v_perm_b32
__device__ __forceinline__ unsigned pack2_bf16(float lo, float hi) {
  union { float f; unsigned u; } a, b; a.f = lo; b.f = hi;
  return __builtin_amdgcn_perm(b.u, a.u, 0x07060302u);
}

union FragAB { v16bf v; unsigned u[8]; };
union FragC  { v8f   v; float   f[8]; };

// ---------------- WMMA fragment loads (ISA 7.12.2 layouts) ----------------
__device__ __forceinline__ void load_frag_a(FragAB& fa, const unsigned short* sh,
                                            int m0, int lane) {
  const unsigned short* base = sh + (m0 + (lane & 15)) * LDS_STRIDE;
  const int khalf = (lane >> 4) * 8;
#pragma unroll
  for (int v = 0; v < 4; ++v)
    fa.u[v] = *(const unsigned*)(base + khalf + 2 * v);
#pragma unroll
  for (int v = 0; v < 4; ++v)
    fa.u[4 + v] = *(const unsigned*)(base + 16 + khalf + 2 * v);
}
__device__ __forceinline__ void load_frag_b(FragAB& fb, const unsigned short* sh,
                                            int n0, int lane) {
  const unsigned short* base = sh + (n0 + (lane & 15)) * LDS_STRIDE;
  const int khalf = (lane >> 4) * 8;
#pragma unroll
  for (int v = 0; v < 4; ++v)
    fb.u[v] = *(const unsigned*)(base + khalf + 2 * v);
#pragma unroll
  for (int v = 0; v < 4; ++v)
    fb.u[4 + v] = *(const unsigned*)(base + 16 + khalf + 2 * v);
}

// ---------------- TDM: 2-D bf16 tile -> LDS with padded rows --------------
#if HAVE_TDM
__device__ __forceinline__ unsigned lds_addr_of(const void* p) {
  return (unsigned)(unsigned long long)
         (__attribute__((address_space(3))) const char*)(const char*)p;
}
// Loads tile_rows x 32 bf16 elements; source rows are rowlen elements apart.
// DMA inserts 16 B of padding every 64 B -> LDS rows are LDS_STRIDE(40) ushorts.
__device__ __forceinline__ void tdm_load_tile(unsigned lds_byte_addr,
                                              const unsigned short* gsrc,
                                              unsigned rowlen_elems,
                                              unsigned tile_rows) {
  unsigned long long ga = (unsigned long long)(const void*)gsrc;
  uint4v g0;
  g0.x = 1u;                                              // count=1, user mode
  g0.y = lds_byte_addr;                                   // lds_addr
  g0.z = (unsigned)(ga & 0xFFFFFFFFu);                    // global_addr[31:0]
  g0.w = (unsigned)((ga >> 32) & 0x01FFFFFFu) | (2u << 30); // ga[56:32] | type=2

  const unsigned td0 = rowlen_elems;                      // tensor_dim0
  const unsigned td1 = 1u << 20;                          // tensor_dim1 (no OOB)
  int8v g1;
  g1[0] = (int)((1u << 16) |      // data_size = 2 bytes
                (1u << 20) |      // pad_enable
                (3u << 22) |      // pad_interval: every 16 dwords (64 B)
                (3u << 25));      // pad_amount : 4 dwords (16 B)
  g1[1] = (int)((td0 & 0xFFFFu) << 16);                          // dim0[15:0]
  g1[2] = (int)(((td0 >> 16) & 0xFFFFu) | ((td1 & 0xFFFFu) << 16)); // dim0 hi | dim1 lo
  g1[3] = (int)(((td1 >> 16) & 0xFFFFu) | ((unsigned)BK << 16));   // dim1 hi | tile_dim0
  g1[4] = (int)(tile_rows & 0xFFFFu);                            // tile_dim1, tile_dim2=0
  g1[5] = (int)rowlen_elems;                                     // dim0_stride[31:0]
  g1[6] = 0;                                                     // stride hi, dim1_stride lo
  g1[7] = 0;
  int4v z4 = {0, 0, 0, 0};
  int8v z8 = {0, 0, 0, 0, 0, 0, 0, 0};
  // clang-23 / therock form: (g0, g1, g2, g3, g4, cpol)
  __builtin_amdgcn_tensor_load_to_lds(g0, g1, z4, z4, z8, 0);
}
#endif

// ---------------- K0a: X f32 -> bf16 --------------------------------------
__global__ __launch_bounds__(256)
void pack_x_bf16(const float* __restrict__ X, unsigned short* __restrict__ Xh) {
  const size_t gid = (size_t)blockIdx.x * 256 + threadIdx.x;
  const float* src = X + gid * 16;
  float4 f0 = ((const float4*)src)[0];
  float4 f1 = ((const float4*)src)[1];
  float4 f2 = ((const float4*)src)[2];
  float4 f3 = ((const float4*)src)[3];
  uint4 o0, o1;
  o0.x = pack2_bf16(f0.x, f0.y); o0.y = pack2_bf16(f0.z, f0.w);
  o0.z = pack2_bf16(f1.x, f1.y); o0.w = pack2_bf16(f1.z, f1.w);
  o1.x = pack2_bf16(f2.x, f2.y); o1.y = pack2_bf16(f2.z, f2.w);
  o1.z = pack2_bf16(f3.x, f3.y); o1.w = pack2_bf16(f3.z, f3.w);
  uint4* dst = (uint4*)(Xh + gid * 16);
  dst[0] = o0; dst[1] = o1;
}

// ---------------- K0b: pack weights ---------------------------------------
__global__ __launch_bounds__(256)
void pack_weights(const float* __restrict__ B_re, const float* __restrict__ B_im,
                  const float* __restrict__ C_re, const float* __restrict__ C_im,
                  const float* __restrict__ D,
                  unsigned short* __restrict__ W1, unsigned short* __restrict__ W2) {
  const int idx = blockIdx.x * 256 + threadIdx.x;
  const int N1 = 1024 * IN_DIM;                  // 262144
  if (idx < N1) {
    const int r = idx >> 8, c = idx & 255;
    const float v = (r < HDIM) ? B_re[r * IN_DIM + c]
                               : B_im[(r - HDIM) * IN_DIM + c];
    W1[idx] = f32_to_bf16_rne(v);
  } else {
    const int j = idx - N1;                      // 0..327679
    const int o = j / 1280, k = j - o * 1280;
    float v;
    if (k < HDIM)            v =  C_re[o * HDIM + k];
    else if (k < 2 * HDIM)   v = -C_im[o * HDIM + (k - HDIM)];
    else                     v =  D[o * IN_DIM + (k - 2 * HDIM)];
    W2[j] = f32_to_bf16_rne(v);
  }
}

// ---------------- K1: input projection (WMMA + TDM) -----------------------
__global__ __launch_bounds__(256)
void lru_proj_gemm(const unsigned short* __restrict__ Xh,
                   const unsigned short* __restrict__ W1,
                   const float* __restrict__ nu_log,
                   const float* __restrict__ bh_re,
                   const float* __restrict__ bh_im,
                   unsigned short* __restrict__ U_re,
                   unsigned short* __restrict__ U_im) {
  __shared__ __align__(16) unsigned short shA[2][BM * LDS_STRIDE];
  __shared__ __align__(16) unsigned short shB[2][BN * LDS_STRIDE];

  const int tid   = threadIdx.x;
  const int lane  = tid & 31;
  const int wave  = tid >> 5;
  const int waveM = wave & 3;
  const int waveN = wave >> 2;

  const int m_base = blockIdx.x * BM;
  const int n_base = blockIdx.y * BN;            // 0..1023 (W1 row space)
  const int isIm   = (n_base >= HDIM) ? 1 : 0;
  const int h_base = n_base & (HDIM - 1);

  FragC acc[2][2];
#pragma unroll
  for (int i = 0; i < 2; ++i)
#pragma unroll
    for (int j = 0; j < 2; ++j)
#pragma unroll
      for (int r = 0; r < 8; ++r) acc[i][j].f[r] = 0.0f;

  const int NK = IN_DIM / BK;                    // 8

#if HAVE_TDM
  const unsigned ldsA[2] = { lds_addr_of(&shA[0][0]), lds_addr_of(&shA[1][0]) };
  const unsigned ldsB[2] = { lds_addr_of(&shB[0][0]), lds_addr_of(&shB[1][0]) };
  const bool issuer = (wave == 0);
  if (issuer) {
    tdm_load_tile(ldsA[0], Xh + (size_t)m_base * IN_DIM, IN_DIM, BM);
    tdm_load_tile(ldsB[0], W1 + (size_t)n_base * IN_DIM, IN_DIM, BN);
  }
  for (int kk = 0; kk < NK; ++kk) {
    if (issuer) __builtin_amdgcn_s_wait_tensorcnt(0);
    __syncthreads();
    const int cur = kk & 1;
    if (issuer && (kk + 1) < NK) {
      const int k1 = (kk + 1) * BK;
      tdm_load_tile(ldsA[cur ^ 1], Xh + (size_t)m_base * IN_DIM + k1, IN_DIM, BM);
      tdm_load_tile(ldsB[cur ^ 1], W1 + (size_t)n_base * IN_DIM + k1, IN_DIM, BN);
    }
    const unsigned short* pA = &shA[cur][0];
    const unsigned short* pB = &shB[cur][0];
#else
  for (int kk = 0; kk < NK; ++kk) {
    const int k0 = kk * BK;
    {  // A tile: 512 x 16B chunks, 2 per thread
#pragma unroll
      for (int i = 0; i < 2; ++i) {
        const int c   = tid + 256 * i;
        const int row = c >> 2, cc = (c & 3) * 8;
        *(uint4*)(&shA[0][row * LDS_STRIDE + cc]) =
            *(const uint4*)(Xh + (size_t)(m_base + row) * IN_DIM + k0 + cc);
      }
      const int row = tid >> 2, cc = (tid & 3) * 8;   // B tile: 1 chunk/thread
      *(uint4*)(&shB[0][row * LDS_STRIDE + cc]) =
          *(const uint4*)(W1 + (size_t)(n_base + row) * IN_DIM + k0 + cc);
    }
    __syncthreads();
    const unsigned short* pA = &shA[0][0];
    const unsigned short* pB = &shB[0][0];
#endif
    FragAB a0, a1, b0, b1;
    load_frag_a(a0, pA, waveM * 32 + 0,  lane);
    load_frag_a(a1, pA, waveM * 32 + 16, lane);
    load_frag_b(b0, pB, waveN * 32 + 0,  lane);
    load_frag_b(b1, pB, waveN * 32 + 16, lane);

    acc[0][0].v = __builtin_amdgcn_wmma_f32_16x16x32_bf16(
        false, a0.v, false, b0.v, (short)0, acc[0][0].v, false, false);
    acc[0][1].v = __builtin_amdgcn_wmma_f32_16x16x32_bf16(
        false, a0.v, false, b1.v, (short)0, acc[0][1].v, false, false);
    acc[1][0].v = __builtin_amdgcn_wmma_f32_16x16x32_bf16(
        false, a1.v, false, b0.v, (short)0, acc[1][0].v, false, false);
    acc[1][1].v = __builtin_amdgcn_wmma_f32_16x16x32_bf16(
        false, a1.v, false, b1.v, (short)0, acc[1][1].v, false, false);
    __syncthreads();
  }

  // Epilogue: u = gamma*(acc + bh) -> bf16 plane.
  unsigned short* Uo = isIm ? U_im : U_re;
#pragma unroll
  for (int j = 0; j < 2; ++j) {
    const int h = h_base + waveN * 32 + j * 16 + (lane & 15);
    const float en    = __expf(nu_log[h]);
    const float gamma = sqrtf(fmaxf(1.0f - __expf(-2.0f * en), 0.0f));
    const float bhv   = isIm ? bh_im[h] : bh_re[h];
#pragma unroll
    for (int i = 0; i < 2; ++i) {
      const int mrow = m_base + waveM * 32 + i * 16 + 8 * (lane >> 4);
#pragma unroll
      for (int r = 0; r < 8; ++r) {
        const float val = gamma * (acc[i][j].f[r] + bhv);
        Uo[(size_t)(mrow + r) * HDIM + h] = f32_to_bf16_trunc(val);
      }
    }
  }
}

// ---------------- K2: serial scan (in place) ------------------------------
__global__ __launch_bounds__(256)
void lru_scan(unsigned short* __restrict__ U_re,
              unsigned short* __restrict__ U_im,
              const float* __restrict__ nu_log,
              const float* __restrict__ theta_log,
              const float* __restrict__ bh_re,
              const float* __restrict__ bh_im,
              float* __restrict__ hN) {
  const int gid = blockIdx.x * blockDim.x + threadIdx.x;  // 0..8191
  const int h = gid & (HDIM - 1);
  const int b = gid >> 9;

  const float en  = __expf(nu_log[h]);
  const float et  = __expf(theta_log[h]);
  const float mag = __expf(-en);
  const float lre = mag * __cosf(et);
  const float lim = mag * __sinf(et);
  const float bre = bh_re[h];
  const float bim = bh_im[h];

  size_t idx = (size_t)b * TLEN * HDIM + h;
  float hre = 0.0f, him = 0.0f;

  for (int t = 0; t < TLEN; ++t, idx += HDIM) {
    if (t + 16 < TLEN) {                        // wave-uniform guard
      __builtin_prefetch(&U_re[idx + 16 * HDIM], 0, 1);
      __builtin_prefetch(&U_im[idx + 16 * HDIM], 0, 1);
    }
    const float ure = bf16_u_to_f32(U_re[idx]);
    const float uim = bf16_u_to_f32(U_im[idx]);
    const float nre = fmaf(lre, hre, fmaf(-lim, him, ure + bre));
    const float nim = fmaf(lre, him, fmaf( lim, hre, uim + bim));
    hre = nre; him = nim;
    U_re[idx] = f32_to_bf16_trunc(hre);
    U_im[idx] = f32_to_bf16_trunc(him);
  }
  hN[2 * gid]     = hre;                        // complex64 interleave
  hN[2 * gid + 1] = him;
}

// ---------------- K3: output projection (WMMA + TDM, K=1280) --------------
__global__ __launch_bounds__(256)
void lru_out_gemm(const unsigned short* __restrict__ YS_re,
                  const unsigned short* __restrict__ YS_im,
                  const unsigned short* __restrict__ Xh,
                  const unsigned short* __restrict__ W2,
                  const float* __restrict__ bias,
                  float* __restrict__ Y) {
  __shared__ __align__(16) unsigned short shA[2][BM * LDS_STRIDE];
  __shared__ __align__(16) unsigned short shB[2][BN * LDS_STRIDE];

  const int tid   = threadIdx.x;
  const int lane  = tid & 31;
  const int wave  = tid >> 5;
  const int waveM = wave & 3;
  const int waveN = wave >> 2;

  const int m_base = blockIdx.x * BM;
  const int n_base = blockIdx.y * BN;            // 0..255

  FragC acc[2][2];
#pragma unroll
  for (int i = 0; i < 2; ++i)
#pragma unroll
    for (int j = 0; j < 2; ++j)
#pragma unroll
      for (int r = 0; r < 8; ++r) acc[i][j].f[r] = 0.0f;

  const int KTOT = 2 * HDIM + IN_DIM;            // 1280
  const int NK   = KTOT / BK;                    // 40

  // A-tile source for K-chunk (segment-uniform: 512/1024 are BK multiples)
  auto a_src = [&](int k0) -> const unsigned short* {
    if (k0 < HDIM)          return YS_re + (size_t)m_base * HDIM + k0;
    if (k0 < 2 * HDIM)      return YS_im + (size_t)m_base * HDIM + (k0 - HDIM);
    return Xh + (size_t)m_base * IN_DIM + (k0 - 2 * HDIM);
  };
  auto a_rowlen = [&](int k0) -> unsigned {
    return (k0 < 2 * HDIM) ? (unsigned)HDIM : (unsigned)IN_DIM;
  };

#if HAVE_TDM
  const unsigned ldsA[2] = { lds_addr_of(&shA[0][0]), lds_addr_of(&shA[1][0]) };
  const unsigned ldsB[2] = { lds_addr_of(&shB[0][0]), lds_addr_of(&shB[1][0]) };
  const bool issuer = (wave == 0);
  if (issuer) {
    tdm_load_tile(ldsA[0], a_src(0), a_rowlen(0), BM);
    tdm_load_tile(ldsB[0], W2 + (size_t)n_base * KTOT, KTOT, BN);
  }
  for (int kk = 0; kk < NK; ++kk) {
    if (issuer) __builtin_amdgcn_s_wait_tensorcnt(0);
    __syncthreads();
    const int cur = kk & 1;
    if (issuer && (kk + 1) < NK) {
      const int k1 = (kk + 1) * BK;
      tdm_load_tile(ldsA[cur ^ 1], a_src(k1), a_rowlen(k1), BM);
      tdm_load_tile(ldsB[cur ^ 1], W2 + (size_t)n_base * KTOT + k1, KTOT, BN);
    }
    const unsigned short* pA = &shA[cur][0];
    const unsigned short* pB = &shB[cur][0];
#else
  for (int kk = 0; kk < NK; ++kk) {
    const int k0 = kk * BK;
    {
      const unsigned short* asrc = a_src(k0);
      const unsigned rl = a_rowlen(k0);
#pragma unroll
      for (int i = 0; i < 2; ++i) {
        const int c   = tid + 256 * i;
        const int row = c >> 2, cc = (c & 3) * 8;
        *(uint4*)(&shA[0][row * LDS_STRIDE + cc]) =
            *(const uint4*)(asrc + (size_t)row * rl + cc);
      }
      const int row = tid >> 2, cc = (tid & 3) * 8;
      *(uint4*)(&shB[0][row * LDS_STRIDE + cc]) =
          *(const uint4*)(W2 + (size_t)(n_base + row) * KTOT + k0 + cc);
    }
    __syncthreads();
    const unsigned short* pA = &shA[0][0];
    const unsigned short* pB = &shB[0][0];
#endif
    FragAB a0, a1, b0, b1;
    load_frag_a(a0, pA, waveM * 32 + 0,  lane);
    load_frag_a(a1, pA, waveM * 32 + 16, lane);
    load_frag_b(b0, pB, waveN * 32 + 0,  lane);
    load_frag_b(b1, pB, waveN * 32 + 16, lane);

    acc[0][0].v = __builtin_amdgcn_wmma_f32_16x16x32_bf16(
        false, a0.v, false, b0.v, (short)0, acc[0][0].v, false, false);
    acc[0][1].v = __builtin_amdgcn_wmma_f32_16x16x32_bf16(
        false, a0.v, false, b1.v, (short)0, acc[0][1].v, false, false);
    acc[1][0].v = __builtin_amdgcn_wmma_f32_16x16x32_bf16(
        false, a1.v, false, b0.v, (short)0, acc[1][0].v, false, false);
    acc[1][1].v = __builtin_amdgcn_wmma_f32_16x16x32_bf16(
        false, a1.v, false, b1.v, (short)0, acc[1][1].v, false, false);
    __syncthreads();
  }

  // Epilogue: Y = acc + bias (f32).
#pragma unroll
  for (int j = 0; j < 2; ++j) {
    const int oc = n_base + waveN * 32 + j * 16 + (lane & 15);
    const float bv = bias[oc];
#pragma unroll
    for (int i = 0; i < 2; ++i) {
      const int mrow = m_base + waveM * 32 + i * 16 + 8 * (lane >> 4);
#pragma unroll
      for (int r = 0; r < 8; ++r)
        Y[(size_t)(mrow + r) * OUT_DIM + oc] = acc[i][j].f[r] + bv;
    }
  }
}

// ---------------------------------------------------------------------------
extern "C" void kernel_launch(void* const* d_in, const int* in_sizes, int n_in,
                              void* d_out, int out_size, void* d_ws, size_t ws_size,
                              hipStream_t stream) {
  const float* X         = (const float*)d_in[0];
  const float* nu_log    = (const float*)d_in[1];
  const float* theta_log = (const float*)d_in[2];
  const float* B_re      = (const float*)d_in[3];
  const float* B_im      = (const float*)d_in[4];
  const float* C_re      = (const float*)d_in[5];
  const float* C_im      = (const float*)d_in[6];
  const float* D         = (const float*)d_in[7];
  const float* bh_re     = (const float*)d_in[8];
  const float* bh_im     = (const float*)d_in[9];
  const float* bias      = (const float*)d_in[10];

  float* Y  = (float*)d_out;
  float* hN = Y + (size_t)MROWS * OUT_DIM;       // h_N complex64 interleaved

  unsigned short* U_re = (unsigned short*)d_ws;                  // 64 MB
  unsigned short* U_im = U_re + (size_t)MROWS * HDIM;            // 64 MB
  unsigned short* Xh   = U_im + (size_t)MROWS * HDIM;            // 32 MB
  unsigned short* W1   = Xh   + (size_t)MROWS * IN_DIM;          // 0.5 MB
  unsigned short* W2   = W1   + (size_t)1024 * IN_DIM;           // 0.64 MB

  dim3 blk(256);
  pack_x_bf16<<<dim3((MROWS * IN_DIM) / (256 * 16)), blk, 0, stream>>>(X, Xh);
  pack_weights<<<dim3((1024 * IN_DIM + 256 * 1280) / 256), blk, 0, stream>>>(
      B_re, B_im, C_re, C_im, D, W1, W2);
  lru_proj_gemm<<<dim3(MROWS / BM, (2 * HDIM) / BN), blk, 0, stream>>>(
      Xh, W1, nu_log, bh_re, bh_im, U_re, U_im);
  lru_scan<<<dim3((BATCH * HDIM) / 256), blk, 0, stream>>>(
      U_re, U_im, nu_log, theta_log, bh_re, bh_im, hN);
  lru_out_gemm<<<dim3(MROWS / BM, OUT_DIM / BN), blk, 0, stream>>>(
      U_re, U_im, Xh, W2, bias, Y);
}
